// HGAT_17901423690205
// MI455X (gfx1250) — matrix operand order repaired and checked by reading, not compiled, not run on hardware
//
#include <hip/hip_runtime.h>

// ---------------------------------------------------------------------------
// HGAT forward for MI455X (gfx1250, wave32, WMMA).
// All GEMMs run on v_wmma_f32_16x16x32_bf16 (fp32 -> bf16 at LDS staging,
// fp32 accumulate). Block tile 64x128, 8 waves, each wave owns a 32x32
// sub-tile (4 accumulators). Staging converts to packed bf16 pairs at load
// time (uint2 per quad: minimal live registers, store phase is ds-only) and
// is software-pipelined one K-chunk ahead so HBM latency overlaps WMMA.
// ---------------------------------------------------------------------------

typedef __attribute__((ext_vector_type(16))) __bf16 v16bf;
typedef __attribute__((ext_vector_type(8)))  float  v8f;

#define BB   48
#define NN   256
#define EE   1024
#define DD   256
#define DIN  256
#define BM   64     // block tile rows
#define BN   128    // block tile cols
#define KC   32     // bf16 wmma K depth
#define STR  36     // padded LDS row stride (bf16 elems, even)
#define LRA  0.2f

union FragBF { v16bf v; unsigned u[8]; };

// two floats -> packed bf16x2 dword (round-to-nearest ties-away):
// 2x v_add + 1x v_perm_b32
__device__ __forceinline__ unsigned pack_bf16x2(float lo, float hi) {
  unsigned ul = __float_as_uint(lo) + 0x8000u;
  unsigned uh = __float_as_uint(hi) + 0x8000u;
  return __builtin_amdgcn_perm(uh, ul, 0x07060302u);  // {uh.hi16, ul.hi16}
}
__device__ __forceinline__ uint2 cvt_quad(float4 v) {
  return make_uint2(pack_bf16x2(v.x, v.y), pack_bf16x2(v.z, v.w));
}

__device__ __forceinline__ float lrelu(float x) { return x >= 0.f ? x : LRA * x; }

__device__ __forceinline__ float wred_sum(float v) {
  #pragma unroll
  for (int o = 16; o > 0; o >>= 1) v += __shfl_xor(v, o, 32);
  return v;
}
__device__ __forceinline__ float wred_max(float v) {
  #pragma unroll
  for (int o = 16; o > 0; o >>= 1) v = fmaxf(v, __shfl_xor(v, o, 32));
  return v;
}

// A fragment: 16 rows x KC from LDS tile laid out [m][k], stride STR.
// ISA layout: lanes 0-15: M=lane, K {0..7}(v0-3),{16..23}(v4-7);
// lanes 16-31: M=lane-16, K {8..15},{24..31}. Pairs packed per dword.
__device__ __forceinline__ v16bf frag_A(const unsigned short* t, int m0, int lane) {
  const unsigned* t32 = (const unsigned*)t;
  int base = (m0 + (lane & 15)) * STR + ((lane < 16) ? 0 : 8);
  FragBF f;
  #pragma unroll
  for (int j = 0; j < 4; ++j) f.u[j]     = t32[(base +      2 * j) >> 1];
  #pragma unroll
  for (int j = 0; j < 4; ++j) f.u[4 + j] = t32[(base + 16 + 2 * j) >> 1];
  return f.v;
}
// B fragment: KC x 16 cols from LDS tile stored TRANSPOSED [n][k], stride STR.
// ISA layout: N = lane%16; lanes 0-15 hold K {0..7},{16..23}; lanes 16-31 K {8..15},{24..31}.
__device__ __forceinline__ v16bf frag_B(const unsigned short* t, int n0, int lane) {
  const unsigned* t32 = (const unsigned*)t;
  int base = (n0 + (lane & 15)) * STR + ((lane < 16) ? 0 : 8);
  FragBF f;
  #pragma unroll
  for (int j = 0; j < 8; ++j) f.u[j] = t32[(base + 2 * (j & 3) + 16 * (j >> 2)) >> 1];
  return f.v;
}
// transpose-store one packed quad (4 consecutive tile rows, same col):
// lowers to ds_store_b16 / ds_store_b16_d16_hi, no extra VALU
__device__ __forceinline__ void stq_T(unsigned short* t, int r0, int c, uint2 q) {
  t[(r0 + 0) * STR + c] = (unsigned short)(q.x);
  t[(r0 + 1) * STR + c] = (unsigned short)(q.x >> 16);
  t[(r0 + 2) * STR + c] = (unsigned short)(q.y);
  t[(r0 + 3) * STR + c] = (unsigned short)(q.y >> 16);
}
// k-contiguous store of one packed quad (one row, 4 consecutive k) as b64
__device__ __forceinline__ void stq_K(unsigned short* t, int r, int c0, uint2 q) {
  *(uint2*)&t[r * STR + c0] = q;
}

#define WMMA_BF16(A, Bv, C) \
  __builtin_amdgcn_wmma_f32_16x16x32_bf16(false, A, false, Bv, (short)0, C, false, false)

// ---------------------------------------------------------------------------
// K0: rdeg[e] = 1 / sum_n inc[n][e]
__global__ __launch_bounds__(256) void k_rdeg(const float* __restrict__ inc,
                                              float* __restrict__ rdeg) {
  int e = blockIdx.x * 256 + threadIdx.x;
  float s = 0.f;
  for (int n = 0; n < NN; ++n) s += inc[n * EE + e];
  rdeg[e] = 1.0f / s;
}

// ---------------------------------------------------------------------------
// K1: h[b][n][d] = sum_c x_in[b][c][n] * W[c][d] + bias[n]
__global__ __launch_bounds__(256) void k_gemm_h(
    const float* __restrict__ x, const float* __restrict__ xm,
    const float* __restrict__ xh, const float* __restrict__ w,
    const float* __restrict__ bias, float* __restrict__ h) {
  __shared__ __align__(16) unsigned short As[BM * STR];
  __shared__ __align__(16) unsigned short Bs[BN * STR];
  const int bb = blockIdx.z;
  const int n0 = blockIdx.y * BM, d0 = blockIdx.x * BN;
  const int tid = threadIdx.x, lane = tid & 31, wave = tid >> 5;
  const int wm = (wave & 1) * 32, wn = (wave >> 1) * 32;
  v8f acc00 = {}, acc01 = {}, acc10 = {}, acc11 = {};

  uint2 qa[2], qb[4];
  const float* pw = w + d0;                         // advances KC*DD per chunk
  auto load_stage = [&](int kc) {
    #pragma unroll
    for (int j = 0; j < 2; ++j) {                   // A: [n][c], src contiguous n
      int q = tid + 256 * j, nq = q & 15, cl = q >> 4;
      int c = kc + cl;
      const float* src; int base;
      if (c < 96)       { src = x;  base = (bb * 96 + c)       * NN; }
      else if (c < 192) { src = xm; base = (bb * 96 + c - 96)  * NN; }
      else              { src = xh; base = (bb * 64 + c - 192) * NN; }
      qa[j] = cvt_quad(*(const float4*)&src[base + n0 + nq * 4]);
    }
    #pragma unroll
    for (int j = 0; j < 4; ++j) {                   // B: [d][c], src contiguous d
      int q = tid + 256 * j, dq = q & 31, cl = q >> 5;
      qb[j] = cvt_quad(*(const float4*)&pw[cl * DD + dq * 4]);
    }
    pw += KC * DD;
  };
  auto store_stage = [&]() {
    #pragma unroll
    for (int j = 0; j < 2; ++j) {
      int q = tid + 256 * j, nq = q & 15, cl = q >> 4;
      stq_T(As, nq * 4, cl, qa[j]);
    }
    #pragma unroll
    for (int j = 0; j < 4; ++j) {
      int q = tid + 256 * j, dq = q & 31, cl = q >> 5;
      stq_T(Bs, dq * 4, cl, qb[j]);
    }
  };

  load_stage(0);
  for (int kc = 0; kc < DIN; kc += KC) {
    store_stage();
    __syncthreads();
    if (kc + KC < DIN) load_stage(kc + KC);         // pipeline next chunk
    v16bf a0 = frag_A(As, wm, lane),      a1 = frag_A(As, wm + 16, lane);
    v16bf b0 = frag_B(Bs, wn, lane),      b1 = frag_B(Bs, wn + 16, lane);
    acc00 = WMMA_BF16(a0, b0, acc00);  acc01 = WMMA_BF16(a0, b1, acc01);
    acc10 = WMMA_BF16(a1, b0, acc10);  acc11 = WMMA_BF16(a1, b1, acc11);
    __syncthreads();
  }
  const int mb = (lane < 16 ? 0 : 8), nl = lane & 15;
  #pragma unroll
  for (int i = 0; i < 2; ++i) {
    v8f A0 = i ? acc10 : acc00, A1 = i ? acc11 : acc01;
    #pragma unroll
    for (int r = 0; r < 8; ++r) {
      int n = n0 + wm + 16 * i + mb + r, d = d0 + wn + nl;
      float bi = bias[n];
      h[(bb * NN + n) * DD + d]      = A0[r] + bi;
      h[(bb * NN + n) * DD + d + 16] = A1[r] + bi;
    }
  }
}

// ---------------------------------------------------------------------------
// K2: edge0[b][d][e] = (sum_n h[b][n][d] * inc[n][e]) * rdeg[e]
__global__ __launch_bounds__(256) void k_gemm_edge0(
    const float* __restrict__ h, const float* __restrict__ inc,
    const float* __restrict__ rdeg, float* __restrict__ edge0) {
  __shared__ __align__(16) unsigned short As[BM * STR];
  __shared__ __align__(16) unsigned short Bs[BN * STR];
  const int bb = blockIdx.z;
  const int dM0 = blockIdx.y * BM, e0 = blockIdx.x * BN;
  const int tid = threadIdx.x, lane = tid & 31, wave = tid >> 5;
  const int wm = (wave & 1) * 32, wn = (wave >> 1) * 32;
  v8f acc00 = {}, acc01 = {}, acc10 = {}, acc11 = {};

  uint2 qa[2], qb[4];
  const float* ph = h + (bb * NN) * DD + dM0;       // + KC*DD per chunk
  const float* pi = inc + e0;                       // + KC*EE per chunk
  auto load_stage = [&](int kc) {
    (void)kc;
    #pragma unroll
    for (int j = 0; j < 2; ++j) {                   // A: [d][n], src h contiguous d
      int q = tid + 256 * j, dq = q & 15, nk = q >> 4;
      qa[j] = cvt_quad(*(const float4*)&ph[nk * DD + dq * 4]);
    }
    #pragma unroll
    for (int j = 0; j < 4; ++j) {                   // B: [e][n], src inc contiguous e
      int q = tid + 256 * j, eq = q & 31, nk = q >> 5;
      qb[j] = cvt_quad(*(const float4*)&pi[nk * EE + eq * 4]);
    }
    ph += KC * DD;  pi += KC * EE;
  };
  auto store_stage = [&]() {
    #pragma unroll
    for (int j = 0; j < 2; ++j) {
      int q = tid + 256 * j, dq = q & 15, nk = q >> 4;
      stq_T(As, dq * 4, nk, qa[j]);
    }
    #pragma unroll
    for (int j = 0; j < 4; ++j) {
      int q = tid + 256 * j, eq = q & 31, nk = q >> 5;
      stq_T(Bs, eq * 4, nk, qb[j]);
    }
  };

  load_stage(0);
  for (int kc = 0; kc < NN; kc += KC) {
    store_stage();
    __syncthreads();
    if (kc + KC < NN) load_stage(kc + KC);
    v16bf a0 = frag_A(As, wm, lane),      a1 = frag_A(As, wm + 16, lane);
    v16bf b0 = frag_B(Bs, wn, lane),      b1 = frag_B(Bs, wn + 16, lane);
    acc00 = WMMA_BF16(a0, b0, acc00);  acc01 = WMMA_BF16(a0, b1, acc01);
    acc10 = WMMA_BF16(a1, b0, acc10);  acc11 = WMMA_BF16(a1, b1, acc11);
    __syncthreads();
  }
  const int mb = (lane < 16 ? 0 : 8), nl = lane & 15;
  #pragma unroll
  for (int i = 0; i < 2; ++i) {
    v8f A0 = i ? acc10 : acc00, A1 = i ? acc11 : acc01;
    #pragma unroll
    for (int r = 0; r < 8; ++r) {
      int d = dM0 + wm + 16 * i + mb + r, e = e0 + wn + nl;
      edge0[(bb * DD + d) * EE + e]      = A0[r] * rdeg[e];
      edge0[(bb * DD + d) * EE + e + 16] = A1[r] * rdeg[e + 16];
    }
  }
}

// ---------------------------------------------------------------------------
// K3: edge[b][e][d] = sum_f edge0[b][d][f] * W2[f][e]   (stored transposed)
__global__ __launch_bounds__(256) void k_gemm_edge(
    const float* __restrict__ edge0, const float* __restrict__ w2,
    float* __restrict__ out_edge) {
  __shared__ __align__(16) unsigned short As[BM * STR];
  __shared__ __align__(16) unsigned short Bs[BN * STR];
  const int bb = blockIdx.z;
  const int dM0 = blockIdx.y * BM, e0 = blockIdx.x * BN;
  const int tid = threadIdx.x, lane = tid & 31, wave = tid >> 5;
  const int wm = (wave & 1) * 32, wn = (wave >> 1) * 32;
  v8f acc00 = {}, acc01 = {}, acc10 = {}, acc11 = {};

  uint2 qa[2], qb[4];
  const float* pa = edge0 + (bb * DD + dM0) * EE;   // + KC per chunk
  const float* pb = w2 + e0;                        // + KC*EE per chunk
  auto load_stage = [&](int kc) {
    #pragma unroll
    for (int j = 0; j < 2; ++j) {                   // A: [d][f], src contiguous f (=k)
      int q = tid + 256 * j, fq = q & 7, dl = q >> 3;
      qa[j] = cvt_quad(*(const float4*)&pa[dl * EE + fq * 4]);
    }
    #pragma unroll
    for (int j = 0; j < 4; ++j) {                   // B: [e][f], src w2 contiguous e
      int q = tid + 256 * j, eq = q & 31, fk = q >> 5;
      qb[j] = cvt_quad(*(const float4*)&pb[fk * EE + eq * 4]);
    }
    pa += KC;  pb += KC * EE;
    if (kc + 2 * KC <= EE)                          // warm L2 for next B tile
      __builtin_prefetch(&pb[(tid >> 3) * EE + ((tid & 7) << 4)], 0, 0);
  };
  auto store_stage = [&]() {
    #pragma unroll
    for (int j = 0; j < 2; ++j) {
      int q = tid + 256 * j, fq = q & 7, dl = q >> 3;
      stq_K(As, dl, fq * 4, qa[j]);                 // packed b64 store
    }
    #pragma unroll
    for (int j = 0; j < 4; ++j) {
      int q = tid + 256 * j, eq = q & 31, fk = q >> 5;
      stq_T(Bs, eq * 4, fk, qb[j]);
    }
  };

  load_stage(0);
  for (int kc = 0; kc < EE; kc += KC) {
    store_stage();
    __syncthreads();
    if (kc + KC < EE) load_stage(kc + KC);
    v16bf a0 = frag_A(As, wm, lane),      a1 = frag_A(As, wm + 16, lane);
    v16bf b0 = frag_B(Bs, wn, lane),      b1 = frag_B(Bs, wn + 16, lane);
    acc00 = WMMA_BF16(a0, b0, acc00);  acc01 = WMMA_BF16(a0, b1, acc01);
    acc10 = WMMA_BF16(a1, b0, acc10);  acc11 = WMMA_BF16(a1, b1, acc11);
    __syncthreads();
  }
  const int mb = (lane < 16 ? 0 : 8), nl = lane & 15;
  #pragma unroll
  for (int i = 0; i < 2; ++i) {
    v8f A0 = i ? acc10 : acc00, A1 = i ? acc11 : acc01;
    #pragma unroll
    for (int r = 0; r < 8; ++r) {
      int d = dM0 + wm + 16 * i + mb + r, e = e0 + wn + nl;
      out_edge[(bb * EE + e)      * DD + d] = A0[r];
      out_edge[(bb * EE + e + 16) * DD + d] = A1[r];
    }
  }
}

// ---------------------------------------------------------------------------
// K4: node_sc[b][n] = sum_d h[b][n][d] * a[d]     (one wave per row)
__global__ __launch_bounds__(256) void k_nodesc(const float* __restrict__ h,
                                                const float* __restrict__ a,
                                                float* __restrict__ node_sc) {
  int lane = threadIdx.x & 31;
  int row = blockIdx.x * 8 + (threadIdx.x >> 5);        // b*NN + n
  const float* hr = h + row * DD;
  float s = 0.f;
  for (int d = lane * 4; d < DD; d += 128) {
    float4 hv  = *(const float4*)&hr[d];
    float4 avv = *(const float4*)&a[d];
    s += hv.x * avv.x + hv.y * avv.y + hv.z * avv.z + hv.w * avv.w;
  }
  s = wred_sum(s);
  if (lane == 0) node_sc[row] = s;
}

// ---------------------------------------------------------------------------
// K5: per (b,e): edge_sc, time_e, softmax column max and 1/sum (wave per col)
__global__ __launch_bounds__(256) void k_edgestats(
    const float* __restrict__ edge, const float* __restrict__ inc,
    const float* __restrict__ pri_e, const float* __restrict__ a,
    const float* __restrict__ a2, const float* __restrict__ a3,
    const float* __restrict__ node_sc, float* __restrict__ edge_sc,
    float* __restrict__ time_e, float* __restrict__ colmax,
    float* __restrict__ colrsum) {
  int lane = threadIdx.x & 31;
  int row = blockIdx.x * 8 + (threadIdx.x >> 5);        // b*EE + e
  int b = row >> 10, e = row & (EE - 1);
  const float* ed = edge + row * DD;
  float se = 0.f, st = 0.f;
  for (int d = lane * 4; d < DD; d += 128) {
    float4 v   = *(const float4*)&ed[d];
    float4 ae  = *(const float4*)&a[DD + d];
    float4 a2v = *(const float4*)&a2[d];
    se += v.x * ae.x  + v.y * ae.y  + v.z * ae.z  + v.w * ae.w;
    st += v.x * a2v.x + v.y * a2v.y + v.z * a2v.z + v.w * a2v.w;
  }
  se = wred_sum(se);
  st = wred_sum(st);
  float te = lrelu(st + pri_e[row] * a2[DD]);
  float a30 = a3[0], a31 = a3[1];
  const float* nsc = node_sc + b * NN;
  float mx = -3.0e38f;
  for (int n = lane; n < NN; n += 32)
    if (inc[n * EE + e] > 0.f)
      mx = fmaxf(mx, lrelu(te * a30 + lrelu(nsc[n] + se) * a31));
  mx = wred_max(mx);
  float sm = 0.f;
  for (int n = lane; n < NN; n += 32)
    if (inc[n * EE + e] > 0.f)
      sm += __expf(lrelu(te * a30 + lrelu(nsc[n] + se) * a31) - mx);
  sm = wred_sum(sm);
  if (lane == 0) {
    edge_sc[row] = se; time_e[row] = te;
    colmax[row] = mx;  colrsum[row] = 1.0f / sm;
  }
}

// ---------------------------------------------------------------------------
// K7: node[b][d][n] = sum_e att[n][e] * edge[e][d]; att regenerated in A-tile
//     staging, 1/sum folded into B-tile.
__global__ __launch_bounds__(256) void k_gemm_node(
    const float* __restrict__ inc, const float* __restrict__ edge,
    const float* __restrict__ node_sc, const float* __restrict__ edge_sc,
    const float* __restrict__ time_e, const float* __restrict__ colmax,
    const float* __restrict__ colrsum, const float* __restrict__ a3,
    float* __restrict__ out_node) {
  __shared__ __align__(16) unsigned short As[BM * STR];
  __shared__ __align__(16) unsigned short Bs[BN * STR];
  const int bb = blockIdx.z;
  const int n0 = blockIdx.y * BM, d0 = blockIdx.x * BN;
  const int tid = threadIdx.x, lane = tid & 31, wave = tid >> 5;
  const int wm = (wave & 1) * 32, wn = (wave >> 1) * 32;
  const float a30 = a3[0], a31 = a3[1];
  v8f acc00 = {}, acc01 = {}, acc10 = {}, acc11 = {};

  uint2 qa[2], qb[4];
  const float* pinc = inc;                          // + KC per chunk
  const float* pte  = time_e  + bb * EE;            // + KC per chunk
  const float* pes  = edge_sc + bb * EE;
  const float* pcm  = colmax  + bb * EE;
  const float* prs  = colrsum + bb * EE;
  const float* pedge = edge + (bb * EE) * DD + d0;  // + KC*DD per chunk
  auto load_stage = [&](int kc) {
    (void)kc;
    #pragma unroll
    for (int j = 0; j < 2; ++j) {                   // A: [n][e], e (=k) contiguous
      int q = tid + 256 * j, eq = q & 7, nl = q >> 3;
      int n = n0 + nl, e4 = eq * 4;
      float4 iv = *(const float4*)&pinc[n * EE + e4];
      float4 tv = *(const float4*)&pte[e4];
      float4 sv = *(const float4*)&pes[e4];
      float4 cv = *(const float4*)&pcm[e4];
      float  nv = node_sc[bb * NN + n];
      float4 p;
      p.x = iv.x > 0.f ? __expf(lrelu(tv.x * a30 + lrelu(nv + sv.x) * a31) - cv.x) : 0.f;
      p.y = iv.y > 0.f ? __expf(lrelu(tv.y * a30 + lrelu(nv + sv.y) * a31) - cv.y) : 0.f;
      p.z = iv.z > 0.f ? __expf(lrelu(tv.z * a30 + lrelu(nv + sv.z) * a31) - cv.z) : 0.f;
      p.w = iv.w > 0.f ? __expf(lrelu(tv.w * a30 + lrelu(nv + sv.w) * a31) - cv.w) : 0.f;
      qa[j] = cvt_quad(p);
    }
    #pragma unroll
    for (int j = 0; j < 4; ++j) {                   // B: [d][e], src edge contiguous d
      int q = tid + 256 * j, dq = q & 31, ek = q >> 5;
      float4 s = *(const float4*)&pedge[ek * DD + dq * 4];
      float  r = prs[ek];
      s.x *= r; s.y *= r; s.z *= r; s.w *= r;
      qb[j] = cvt_quad(s);
    }
    pinc += KC; pte += KC; pes += KC; pcm += KC; prs += KC;
    pedge += KC * DD;
  };
  auto store_stage = [&]() {
    #pragma unroll
    for (int j = 0; j < 2; ++j) {
      int q = tid + 256 * j, eq = q & 7, nl = q >> 3;
      stq_K(As, nl, eq * 4, qa[j]);                 // packed b64 store
    }
    #pragma unroll
    for (int j = 0; j < 4; ++j) {
      int q = tid + 256 * j, dq = q & 31, ek = q >> 5;
      stq_T(Bs, dq * 4, ek, qb[j]);
    }
  };

  load_stage(0);
  for (int kc = 0; kc < EE; kc += KC) {
    store_stage();
    __syncthreads();
    if (kc + KC < EE) load_stage(kc + KC);
    v16bf a0 = frag_A(As, wm, lane),      a1 = frag_A(As, wm + 16, lane);
    v16bf b0 = frag_B(Bs, wn, lane),      b1 = frag_B(Bs, wn + 16, lane);
    acc00 = WMMA_BF16(a0, b0, acc00);  acc01 = WMMA_BF16(a0, b1, acc01);
    acc10 = WMMA_BF16(a1, b0, acc10);  acc11 = WMMA_BF16(a1, b1, acc11);
    __syncthreads();
  }
  const int mb = (lane < 16 ? 0 : 8), nl = lane & 15;
  #pragma unroll
  for (int i = 0; i < 2; ++i) {
    v8f A0 = i ? acc10 : acc00, A1 = i ? acc11 : acc01;
    #pragma unroll
    for (int r = 0; r < 8; ++r) {
      int n = n0 + wm + 16 * i + mb + r, d = d0 + wn + nl;   // out is [b][d][n]
      out_node[(bb * DD + d)      * NN + n] = A0[r];
      out_node[(bb * DD + d + 16) * NN + n] = A1[r];
    }
  }
}

// ---------------------------------------------------------------------------
extern "C" void kernel_launch(void* const* d_in, const int* in_sizes, int n_in,
                              void* d_out, int out_size, void* d_ws, size_t ws_size,
                              hipStream_t stream) {
  (void)in_sizes; (void)n_in; (void)out_size; (void)ws_size;
  const float* x     = (const float*)d_in[0];
  const float* xm    = (const float*)d_in[1];
  const float* xh    = (const float*)d_in[2];
  const float* pri_e = (const float*)d_in[3];
  // d_in[4] = pri_n: unused by the forward pass (faithful to reference)
  const float* inc   = (const float*)d_in[5];
  const float* w     = (const float*)d_in[6];
  const float* bias  = (const float*)d_in[7];
  const float* w2    = (const float*)d_in[8];
  const float* a     = (const float*)d_in[9];
  const float* a2    = (const float*)d_in[10];
  const float* a3    = (const float*)d_in[11];

  float* out_node = (float*)d_out;                        // [B][D][N]
  float* out_edge = out_node + BB * DD * NN;              // [B][E][D]

  float* ws      = (float*)d_ws;
  float* h       = ws;                                    // B*N*D
  float* edge0   = h + BB * NN * DD;                      // B*D*E
  float* rdeg    = edge0 + BB * DD * EE;                  // E
  float* node_sc = rdeg + EE;                             // B*N
  float* edge_sc = node_sc + BB * NN;                     // B*E
  float* time_e  = edge_sc + BB * EE;                     // B*E
  float* colmax  = time_e + BB * EE;                      // B*E
  float* colrsum = colmax + BB * EE;                      // B*E

  k_rdeg      <<<EE / 256, 256, 0, stream>>>(inc, rdeg);
  k_gemm_h    <<<dim3(DD / BN, NN / BM, BB), 256, 0, stream>>>(x, xm, xh, w, bias, h);
  k_gemm_edge0<<<dim3(EE / BN, DD / BM, BB), 256, 0, stream>>>(h, inc, rdeg, edge0);
  k_gemm_edge <<<dim3(EE / BN, DD / BM, BB), 256, 0, stream>>>(edge0, w2, out_edge);
  k_nodesc    <<<BB * NN / 8, 256, 0, stream>>>(h, a, node_sc);
  k_edgestats <<<BB * EE / 8, 256, 0, stream>>>(out_edge, inc, pri_e, a, a2, a3,
                                                node_sc, edge_sc, time_e, colmax, colrsum);
  k_gemm_node <<<dim3(DD / BN, NN / BM, BB), 256, 0, stream>>>(inc, out_edge, node_sc,
                                                edge_sc, time_e, colmax, colrsum, a3, out_node);
}
